// RNNModel_37855841747123
// MI455X (gfx1250) — compile-verified, matrix operand order checked by reading, not run on hardware
//
#include <hip/hip_runtime.h>

// Problem constants (match reference).
#define Bb    64
#define Ss    256
#define Ee    512
#define Hh    1024
#define Dd    512
#define Cc    2
#define G4H   4096          // 4*H
#define KS_E  16            // Ee/32 k-steps over embedding part
#define KS_H  32            // Hh/32 k-steps over recurrent part
#define KS_T  48            // total k-steps (K = 1536)
#define NBLK  64            // persistent grid (co-resident on MI455X)

// Dynamic LDS: 4 gate weight tiles (4*48*64*16 B) + gate spill buffer.
#define WLDS_QUADS (4 * KS_T * 64)               // 12288 x 16 B = 196608 B
#define GLDS_OFF   (WLDS_QUADS * 16)
#define SMEM_BYTES (GLDS_OFF + 4 * 64 * 16 * 4)  // + 16384 = 212992 B

typedef __attribute__((ext_vector_type(16))) __bf16       v16bf;
typedef __attribute__((ext_vector_type(8)))  float        v8f;
typedef __attribute__((ext_vector_type(4)))  unsigned int u32x4;

union Frag { v16bf v; u32x4 q[2]; };

__device__ __forceinline__ unsigned short f2bf(float f) {
  union { float f; unsigned u; } v; v.f = f;
  unsigned r = v.u + 0x7FFFu + ((v.u >> 16) & 1u);   // round-to-nearest-even
  return (unsigned short)(r >> 16);
}
__device__ __forceinline__ float sigm(float x) { return 1.0f / (1.0f + expf(-x)); }

// ---------------------------------------------------------------------------
// Pack lstm_kernel (f32 [1536][4096]) -> bf16 fragment-major B tiles:
//   [ntile(256)][ks(48)][chunk(2)][lane(32)][8 bf16]
// ---------------------------------------------------------------------------
__global__ __launch_bounds__(256) void pack_weights_kernel(
    const float* __restrict__ w, unsigned short* __restrict__ dst) {
  int g = blockIdx.x * 256 + threadIdx.x;          // 786432 groups of 8
  int lane  = g & 31;
  int ch    = (g >> 5) & 1;
  int ks    = (g >> 6) % KS_T;
  int ntile = (g >> 6) / KS_T;
  int n     = ntile * 16 + (lane & 15);
  int half  = lane >> 4;
  int kbase = ks * 32 + ch * 16 + half * 8;
  unsigned short* d = dst + (size_t)g * 8;
#pragma unroll
  for (int j = 0; j < 8; ++j)
    d[j] = f2bf(w[(size_t)(kbase + j) * G4H + n]);
}

// ---------------------------------------------------------------------------
// Gather embeddings and pack into fragment-major A tiles:
//   [s(256)][rt(4)][ks(16)][chunk(2)][lane(32)][8 bf16]   (row m = batch)
// ---------------------------------------------------------------------------
__global__ __launch_bounds__(256) void pack_emb_kernel(
    const int* __restrict__ X, const float* __restrict__ table,
    unsigned short* __restrict__ dst) {
  int g = blockIdx.x * 256 + threadIdx.x;          // 1048576 groups of 8
  int lane = g & 31;
  int ch   = (g >> 5) & 1;
  int ks   = (g >> 6) & 15;
  int rt   = (g >> 10) & 3;
  int s    = g >> 12;
  int b    = rt * 16 + (lane & 15);
  int half = lane >> 4;
  int kbase = ks * 32 + ch * 16 + half * 8;
  int tok = X[b * Ss + s];
  const float* row = table + (size_t)tok * Ee + kbase;
  unsigned short* d = dst + (size_t)g * 8;
#pragma unroll
  for (int j = 0; j < 8; ++j) d[j] = f2bf(row[j]);
}

__global__ __launch_bounds__(256) void init_state_kernel(
    float* __restrict__ c, unsigned short* __restrict__ h0,
    unsigned* __restrict__ barrier_cnt) {
  int i = blockIdx.x * 256 + threadIdx.x;          // 65536 = B*H
  c[i]  = 0.0f;
  h0[i] = 0;
  if (i == 0) *barrier_cnt = 0u;
}

// Device-scope grid barrier for the persistent LSTM kernel.
__device__ __forceinline__ void grid_barrier(unsigned* cnt, unsigned target) {
  __syncthreads();
  if (threadIdx.x == 0) {
    __builtin_amdgcn_fence(__ATOMIC_RELEASE, "agent");   // flush h/c writes
    __hip_atomic_fetch_add(cnt, 1u, __ATOMIC_RELAXED, __HIP_MEMORY_SCOPE_AGENT);
    while (__hip_atomic_load(cnt, __ATOMIC_RELAXED, __HIP_MEMORY_SCOPE_AGENT) < target)
      __builtin_amdgcn_s_sleep(1);
  }
  __syncthreads();
  __builtin_amdgcn_fence(__ATOMIC_ACQUIRE, "agent");     // invalidate stale lines
}

// ---------------------------------------------------------------------------
// Persistent LSTM: 64 workgroups x 512 threads (16 waves), all 256 timesteps.
// Workgroup wg owns hidden columns [wg*16, wg*16+16).
// Wave w: row-tile rt = w>>2, gate g = w&3 (i,j,f,o); one [16x16] tile, K=1536.
// Weights for the block's 4 gate tiles live in LDS (192 KB of the WGP's 320 KB).
// ---------------------------------------------------------------------------
__global__ __launch_bounds__(512) void lstm_persistent_kernel(
    const u32x4* __restrict__ embP,      // packed emb fragments
    const u32x4* __restrict__ wtP,       // packed weight fragments
    unsigned short* h0,                  // packed h buffer 0 (zero-initialized)
    unsigned short* h1,                  // packed h buffer 1
    float* __restrict__ cState,          // [B][H] f32
    const float* __restrict__ bias,      // [4H]
    const int* __restrict__ seqlen,      // [B]
    float* __restrict__ finalH,          // [B][H] f32
    unsigned* barrier_cnt)
{
  extern __shared__ char smem[];
  u32x4* wLds = (u32x4*)smem;                   // [g(4)][ks(48)][ch(2)][lane(32)]
  float* gLds = (float*)(smem + GLDS_OFF);      // [4][64][16]

  const int wg   = blockIdx.x;           // 16-col hidden tile
  const int tid  = threadIdx.x;
  const int wave = tid >> 5;
  const int lane = tid & 31;
  const int rt   = wave >> 2;            // 0..3
  const int g    = wave & 3;             // gate i/j/f/o
  const int kbase = wg * 16;

  // One-time async copy of this block's weight tiles into LDS (ASYNCcnt path).
  for (int i = tid; i < WLDS_QUADS; i += 512) {
    int gg = i / (KS_T * 64);
    int r  = i % (KS_T * 64);
    const u32x4* src = wtP + ((size_t)(gg * 64 + wg) * (KS_T * 64) + r);
    unsigned dstOff = (unsigned)(size_t)(wLds + i);
    asm volatile("global_load_async_to_lds_b128 %0, %1, off"
                 :: "v"(dstOff), "v"(src) : "memory");
  }
  asm volatile("s_wait_asynccnt 0" ::: "memory");
  __syncthreads();

  const u32x4* bp = wLds + (size_t)g * (KS_T * 64);

  for (int s = 0; s < Ss; ++s) {
    const unsigned short* hPrevRaw = (s & 1) ? h1 : h0;
    unsigned short*       hNext    = (s & 1) ? h0 : h1;
    const u32x4* aE = embP + ((size_t)s * 4 + rt) * (KS_E * 64);
    const u32x4* aH = (const u32x4*)hPrevRaw + (size_t)rt * (KS_H * 64);

    v8f acc = {};
    // K over embedding part (k = 0..511): A from global (L2), B from LDS.
    for (int ks = 0; ks < KS_E; ++ks) {
      Frag a, b;
      a.q[0] = aE[ks * 64 + lane];   a.q[1] = aE[ks * 64 + 32 + lane];
      b.q[0] = bp[ks * 64 + lane];   b.q[1] = bp[ks * 64 + 32 + lane];
      acc = __builtin_amdgcn_wmma_f32_16x16x32_bf16(false, a.v, false, b.v,
                                                    (short)0, acc, false, false);
    }
    // K over recurrent part (k = 512..1535)
    for (int ks = 0; ks < KS_H; ++ks) {
      Frag a, b;
      const int kk = KS_E + ks;
      a.q[0] = aH[ks * 64 + lane];   a.q[1] = aH[ks * 64 + 32 + lane];
      b.q[0] = bp[kk * 64 + lane];   b.q[1] = bp[kk * 64 + 32 + lane];
      acc = __builtin_amdgcn_wmma_f32_16x16x32_bf16(false, a.v, false, b.v,
                                                    (short)0, acc, false, false);
    }

    // Spill gate tile to LDS (D layout: lane = n + 16*(m>=8), vgpr r = m&7)
    {
      const int n  = lane & 15;
      const int mh = lane >> 4;
#pragma unroll
      for (int r = 0; r < 8; ++r)
        gLds[(g * 64 + rt * 16 + mh * 8 + r) * 16 + n] = acc[r];
    }
    __syncthreads();

    // Elementwise LSTM cell update (tf BasicLSTMCell, forget bias 1.0)
#pragma unroll
    for (int it = 0; it < 2; ++it) {
      int idx = tid + it * 512;          // 1024 = 64 rows x 16 cols
      int m = idx >> 4, n = idx & 15;
      int k = kbase + n;
      float gi = gLds[(0 * 64 + m) * 16 + n] + bias[k];
      float gj = gLds[(1 * 64 + m) * 16 + n] + bias[Hh + k];
      float gf = gLds[(2 * 64 + m) * 16 + n] + bias[2 * Hh + k];
      float go = gLds[(3 * 64 + m) * 16 + n] + bias[3 * Hh + k];
      float cp = cState[m * Hh + k];
      float nc = cp * sigm(gf + 1.0f) + sigm(gi) * tanhf(gj);
      float nh = tanhf(nc) * sigm(go);
      cState[m * Hh + k] = nc;
      // store next-h directly in fragment-packed bf16 layout
      int ks2 = k >> 5, r = k & 31;
      int ch = r >> 4, half = (r >> 3) & 1, j = r & 7;
      int lane2 = (m & 15) + 16 * half;
      int rt2 = m >> 4;
      hNext[(size_t)((((rt2 * 32 + ks2) * 2 + ch) * 32 + lane2) * 8 + j)] = f2bf(nh);
      if (s + 1 == seqlen[m]) finalH[m * Hh + k] = nh;
    }

    // Make every block's hNext visible before anyone starts step s+1.
    grid_barrier(barrier_cnt, (unsigned)(NBLK * (s + 1)));
  }
}

// ---------------------------------------------------------------------------
// dense = relu(finalH @ dense_w + dense_b)   [64,1024]@[1024,512]
// ---------------------------------------------------------------------------
__global__ __launch_bounds__(256) void dense_relu_kernel(
    const float* __restrict__ fh, const float* __restrict__ w,
    const float* __restrict__ b, float* __restrict__ out) {
  int idx = blockIdx.x * 256 + threadIdx.x;   // 32768
  int row = idx >> 9, d = idx & 511;
  float acc = b[d];
  const float* fr = fh + (size_t)row * Hh;
  for (int k = 0; k < Hh; ++k) acc += fr[k] * w[(size_t)k * Dd + d];
  out[idx] = fmaxf(acc, 0.0f);
}

// logits = dense @ pred_w + pred_b   [64,512]@[512,2]
__global__ void logits_kernel(
    const float* __restrict__ dn, const float* __restrict__ w,
    const float* __restrict__ b, float* __restrict__ out) {
  int idx = threadIdx.x;                      // 128
  int row = idx >> 1, c = idx & 1;
  float acc = b[c];
  const float* dr = dn + (size_t)row * Dd;
  for (int k = 0; k < Dd; ++k) acc += dr[k] * w[k * Cc + c];
  out[idx] = acc;
}

// ---------------------------------------------------------------------------
extern "C" void kernel_launch(void* const* d_in, const int* in_sizes, int n_in,
                              void* d_out, int out_size, void* d_ws, size_t ws_size,
                              hipStream_t stream) {
  const int*   X       = (const int*)d_in[0];
  const int*   seqlen  = (const int*)d_in[1];
  const float* embT    = (const float*)d_in[2];
  const float* lstm_w  = (const float*)d_in[3];
  const float* lstm_b  = (const float*)d_in[4];
  const float* dense_w = (const float*)d_in[5];
  const float* dense_b = (const float*)d_in[6];
  const float* pred_w  = (const float*)d_in[7];
  const float* pred_b  = (const float*)d_in[8];
  float* out = (float*)d_out;

  char* ws = (char*)d_ws;
  size_t off = 0;
  auto carve = [&](size_t bytes) {
    void* p = ws + off;
    off = (off + bytes + 255) & ~(size_t)255;
    return p;
  };
  unsigned short* wtP  = (unsigned short*)carve((size_t)KS_T * G4H * 16 * 2 * 2); // 12.6 MB
  unsigned short* embP = (unsigned short*)carve((size_t)Bb * Ss * Ee * 2);        // 16.8 MB
  unsigned short* h0   = (unsigned short*)carve((size_t)Bb * Hh * 2);
  unsigned short* h1   = (unsigned short*)carve((size_t)Bb * Hh * 2);
  float*    cS   = (float*)carve((size_t)Bb * Hh * 4);
  float*    fH   = (float*)carve((size_t)Bb * Hh * 4);
  float*    dnO  = (float*)carve((size_t)Bb * Dd * 4);
  unsigned* bCnt = (unsigned*)carve(256);

  // Allow > 64 KB dynamic LDS for the persistent kernel (WGP has 320 KB).
  (void)hipFuncSetAttribute((const void*)lstm_persistent_kernel,
                            hipFuncAttributeMaxDynamicSharedMemorySize,
                            SMEM_BYTES);

  pack_weights_kernel<<<3072, 256, 0, stream>>>(lstm_w, wtP);
  pack_emb_kernel<<<4096, 256, 0, stream>>>(X, embT, embP);
  init_state_kernel<<<256, 256, 0, stream>>>(cS, h0, bCnt);

  lstm_persistent_kernel<<<NBLK, 512, SMEM_BYTES, stream>>>(
      (const u32x4*)embP, (const u32x4*)wtP, h0, h1,
      cS, lstm_b, seqlen, fH, bCnt);

  dense_relu_kernel<<<128, 256, 0, stream>>>(fH, dense_w, dense_b, dnO);
  logits_kernel<<<1, 128, 0, stream>>>(dnO, pred_w, pred_b, out);
}
